// SoftmaxTopKMaxPooling_Decode_Fused_15547781612185
// MI455X (gfx1250) — matrix operand order repaired and checked by reading, not compiled, not run on hardware
//
#include <hip/hip_runtime.h>
#include <hip/hip_bf16.h>

typedef __bf16 v8bf  __attribute__((ext_vector_type(8)));
typedef __bf16 v16bf __attribute__((ext_vector_type(16)));
typedef float  v8f   __attribute__((ext_vector_type(8)));

// Shapes (fixed by the reference setup)
#define BDIM 32
#define HDIM 4
#define GDIM 16
#define DDIM 128
#define SDIM 2048
#define TOPK 64
#define SM_SCALE 0.08838834764831845f  // 1/sqrt(128)
#define PFDIST 4                       // prefetch distance in 16-key tiles (64 KB/wave ahead)

// LDS layout (in floats)
#define SC_STRIDE 2052                       // 2048 + 4 pad -> column reads hit distinct banks
#define SC_SIZE   (GDIM * SC_STRIDE)         // 32832
#define POOL_OFF  SC_SIZE                    // 2048 floats
#define RED_OFF   (POOL_OFF + SDIM)          // byte offset 139520, 8B aligned
#define RED_SIZE  16                         // 8 x u64
#define PART_OFF  (RED_OFF + RED_SIZE)       // 256 floats
#define LSE_OFF   (PART_OFF + 256)           // 16 floats
#define IDX_OFF   (LSE_OFF + 16)             // 64 ints
#define SMEM_FLOATS (IDX_OFF + 64)
#define SMEM_BYTES  (SMEM_FLOATS * 4)        // ~141 KB; 2 blocks/WGP fit in 320 KB

__device__ __forceinline__ unsigned ford(float f) {
    unsigned u = __float_as_uint(f);
    return (u & 0x80000000u) ? ~u : (u | 0x80000000u);
}
__device__ __forceinline__ int ffin(float f) {
    return (__float_as_uint(f) & 0x7F800000u) != 0x7F800000u;
}

__global__ __launch_bounds__(256)
void hsa_decode_fused_kernel(const __hip_bfloat16* __restrict__ Qg,
                             const __hip_bfloat16* __restrict__ Kg,
                             const int* __restrict__ seq_lens,
                             float* __restrict__ out_scores,
                             int* __restrict__ out_idx) {
    extern __shared__ float smem[];
    float* sc      = smem;                                   // [16][2052] scaled raw scores
    float* pooled  = smem + POOL_OFF;                        // [2048]
    unsigned long long* red = (unsigned long long*)(smem + RED_OFF); // [8]
    float* part    = smem + PART_OFF;                        // [16][16]
    float* lse     = smem + LSE_OFF;                         // [16]
    int*   idxl    = (int*)(smem + IDX_OFF);                 // [64]

    const int tid  = threadIdx.x;
    const int bh   = blockIdx.x;           // b*H + h
    const int b    = bh >> 2;
    const int h    = bh & 3;

    const int L = seq_lens[b];
    int nvalid = L / 64;
    int thr    = (L - 512) / 64;
    if (thr < nvalid) nvalid = thr;
    if (nvalid > SDIM) nvalid = SDIM;
    if (nvalid < 0) nvalid = 0;

    // ---------------- Phase 1: scores = (Q K^T) * scale via WMMA ----------------
    const int lane = tid & 31;
    const int wave = tid >> 5;             // 8 waves, each owns 256 keys
    const int hi   = lane >> 4;            // half-wave selector
    const int l15  = lane & 15;

    const __bf16* Qp = reinterpret_cast<const __bf16*>(Qg);
    const __bf16* Kp = reinterpret_cast<const __bf16*>(Kg);

    // A fragments: row g = l15; lanes 0-15 hold d {k*32+0..15},
    // lanes 16-31 hold d {k*32+16..31}  (contiguous k order, matches B)
    const __bf16* Qb = Qp + (((size_t)bh) * GDIM + l15) * DDIM;
    v16bf afrag[8];
#pragma unroll
    for (int kk = 0; kk < 8; ++kk) {
        const __bf16* p = Qb + kk * 32 + hi * 16;
        v8bf lo = *(const v8bf*)p;
        v8bf hh = *(const v8bf*)(p + 8);
        afrag[kk] = __builtin_shufflevector(lo, hh,
            0, 1, 2, 3, 4, 5, 6, 7, 8, 9, 10, 11, 12, 13, 14, 15);
    }

    // Per-lane K base: column n = l15 -> key s; lanes 0-15 take d lo-half, 16-31 hi-half
    const __bf16* Krow  = Kp + (((size_t)b * SDIM + (wave * 256 + l15)) * HDIM + h) * DDIM;
    const __bf16* Kbase = Krow + hi * 16;                    // load base (d-halves)
    const char*   Kpf   = (const char*)Krow + hi * 128;      // prefetch base: 2x128B lines/row

    // Software-pipelined, scheduler-pinned double buffer
    v16bf bbuf[2][8];
#pragma unroll
    for (int kk = 0; kk < 8; ++kk)
        bbuf[0][kk] = *(const v16bf*)(Kbase + kk * 32);
    __builtin_amdgcn_sched_barrier(0);

#pragma unroll
    for (int nt = 0; nt < 16; ++nt) {
        if (nt + 1 < 16) {
            const __bf16* Kn = Kbase + (size_t)(nt + 1) * 16 * HDIM * DDIM;
#pragma unroll
            for (int kk = 0; kk < 8; ++kk)
                bbuf[(nt + 1) & 1][kk] = *(const v16bf*)(Kn + kk * 32);
        }
        __builtin_amdgcn_sched_barrier(0);   // keep next-tile loads ahead of this tile's math

        if (nt + PFDIST < 16)                // stream-prefetch 4 tiles ahead (no counter cost)
            __builtin_prefetch(Kpf + (size_t)(nt + PFDIST) * 16 * HDIM * DDIM * 2, 0, 1);

        v8f acc = {};
#pragma unroll
        for (int kk = 0; kk < 8; ++kk) {
            acc = __builtin_amdgcn_wmma_f32_16x16x32_bf16(
                false, afrag[kk], false, bbuf[nt & 1][kk], (short)0, acc, false, false);
        }
        // C/D layout: vgpr j -> row M = j + 8*hi, col N = l15
        const int s  = wave * 256 + nt * 16 + l15;
        const int g0 = hi * 8;
#pragma unroll
        for (int j = 0; j < 8; ++j)
            sc[(g0 + j) * SC_STRIDE + s] = acc[j] * SM_SCALE;
        __builtin_amdgcn_sched_barrier(0);
    }
    __syncthreads();

    // ---------------- Phase 2: per-row masked LSE ----------------
    const int g  = tid >> 4;
    const int jj = tid & 15;

    float mloc = -INFINITY;
    for (int s = jj; s < nvalid; s += 16)
        mloc = fmaxf(mloc, sc[g * SC_STRIDE + s]);
    part[tid] = mloc;
    __syncthreads();
    if (tid < 16) {
        float mm = -INFINITY;
        for (int i = 0; i < 16; ++i) mm = fmaxf(mm, part[tid * 16 + i]);
        lse[tid] = mm;                                   // raw max, temporarily
    }
    __syncthreads();

    const float mg    = lse[g];
    const float safem = ffin(mg) ? mg : 0.0f;
    float ssum = 0.0f;
    for (int s = jj; s < nvalid; s += 16)
        ssum += expf(sc[g * SC_STRIDE + s] - safem);
    part[tid] = ssum;
    __syncthreads();
    if (tid < 16) {
        float tt = 0.0f;
        for (int i = 0; i < 16; ++i) tt += part[tid * 16 + i];
        const float m2  = lse[tid];
        const float sm2 = ffin(m2) ? m2 : 0.0f;
        lse[tid] = (tt > 0.0f) ? (sm2 + logf(tt)) : -INFINITY;
    }
    __syncthreads();

    // ---------------- Phase 3: pooled[s] = max_g (score - lse) ----------------
    for (int s = tid; s < SDIM; s += 256) {
        float p = -INFINITY;
        if (s < nvalid) {
#pragma unroll
            for (int gg = 0; gg < GDIM; ++gg)
                p = fmaxf(p, sc[gg * SC_STRIDE + s] - lse[gg]);
        }
        pooled[s] = p;
    }
    __syncthreads();

    // ---------------- Phase 4: top-64 (value desc, index asc on ties) ----------------
    // Packed key: (monotone f32 ord) << 32 | (2047 - s) -> max key == max val, min index
    for (int k = 0; k < TOPK; ++k) {
        unsigned long long best = 0ull;
#pragma unroll
        for (int i = 0; i < SDIM / 256; ++i) {
            const int s = tid + i * 256;
            unsigned long long key =
                ((unsigned long long)ford(pooled[s]) << 32) |
                (unsigned)(SDIM - 1 - s);
            if (key > best) best = key;
        }
        // wave-level reduction, no barriers
#pragma unroll
        for (int off = 16; off > 0; off >>= 1) {
            unsigned long long o = __shfl_xor(best, off, 32);
            if (o > best) best = o;
        }
        if (lane == 0) red[wave] = best;
        __syncthreads();
        if (tid == 0) {
            unsigned long long kb = red[0];
#pragma unroll
            for (int i = 1; i < 8; ++i)
                if (red[i] > kb) kb = red[i];
            const int s   = (SDIM - 1) - (int)(kb & 0xFFFFFFFFull);
            const float v = pooled[s];
            idxl[k]   = ffin(v) ? s : -1;
            pooled[s] = -INFINITY;                        // consume
        }
        __syncthreads();
    }

    // ---------------- Phase 5: gather outputs ----------------
    // sel[b,h,g,k] == scores[g, idx_k] (same GEMM), masked where idx < 0
    for (int i = tid; i < GDIM * TOPK; i += 256) {
        const int gg  = i >> 6;
        const int k   = i & 63;
        const int idx = idxl[k];
        out_scores[((size_t)bh * GDIM + gg) * TOPK + k] =
            (idx >= 0) ? sc[gg * SC_STRIDE + idx] : -INFINITY;
    }
    if (tid < TOPK)
        out_idx[bh * TOPK + tid] = idxl[tid];
}

extern "C" void kernel_launch(void* const* d_in, const int* in_sizes, int n_in,
                              void* d_out, int out_size, void* d_ws, size_t ws_size,
                              hipStream_t stream) {
    (void)in_sizes; (void)n_in; (void)d_ws; (void)ws_size; (void)out_size;
    const __hip_bfloat16* Q = (const __hip_bfloat16*)d_in[0];
    const __hip_bfloat16* K = (const __hip_bfloat16*)d_in[1];
    const int* seq_lens     = (const int*)d_in[2];

    float* out_scores = (float*)d_out;                       // (32,4,16,64) f32
    int*   out_idx    = (int*)(out_scores + BDIM * HDIM * GDIM * TOPK); // (32,4,64) i32

    hsa_decode_fused_kernel<<<dim3(BDIM * HDIM), dim3(256), SMEM_BYTES, stream>>>(
        Q, K, seq_lens, out_scores, out_idx);
}
